// TopDown_Bottomup_YsYp_36507222016276
// MI455X (gfx1250) — matrix operand order repaired and checked by reading, not compiled, not run on hardware
//
#include <hip/hip_runtime.h>
#include <cstddef>
#include <cstdint>

// ---------------- problem constants (from reference) ----------------
#define V_N   20000
#define L_N   4
#define PER_N 5000
#define DEG_N 16
#define D_N   512
#define H_N   8
#define CD_N  64
#define PB_N  (PER_N * DEG_N)           // 80000 edges per level block
#define E_N   ((L_N - 1) * PB_N)        // 240000 edges
#define SCALE_A 0.07216878364870322f    // (3*Cd)^-0.5 = 192^-0.5

typedef __attribute__((ext_vector_type(16))) _Float16 v16h;
typedef __attribute__((ext_vector_type(8)))  float    v8f;

union HF16x16 { v16h v; _Float16 h[16]; };

__device__ __forceinline__ int orderF(float f) {
    int i = __float_as_int(f);
    return (i >= 0) ? i : (i ^ 0x7fffffff);
}
__device__ __forceinline__ float unorderF(int i) {
    return __int_as_float((i >= 0) ? i : (i ^ 0x7fffffff));
}

// ---- CDNA5 async copy global->LDS (ASYNCcnt-tracked, bypasses VGPRs) ----
__device__ __forceinline__ void async_copy_b128(unsigned lds_off, const void* gptr) {
    asm volatile("global_load_async_to_lds_b128 %0, %1, off"
                 :: "v"(lds_off), "v"((unsigned long long)(uintptr_t)gptr)
                 : "memory");
}
__device__ __forceinline__ void async_wait0() {
    asm volatile("s_wait_asynccnt 0" ::: "memory");
}

// ---------------- weight convert + transpose to f16 ----------------
// out[n*512 + k] = (f16) W[k*512 + n]
__global__ __launch_bounds__(256) void k_cvt_transpose(const float* __restrict__ W,
                                                       _Float16* __restrict__ out) {
    int idx = blockIdx.x * 256 + threadIdx.x;            // 512*512 threads
    if (idx >= D_N * D_N) return;
    int n = idx & (D_N - 1);
    int k = idx >> 9;
    out[(size_t)n * D_N + k] = (_Float16)W[(size_t)k * D_N + n];
}

// ---------------- fold We with aw_e:  w_alpha[k*8+h] ----------------
__global__ __launch_bounds__(256) void k_fold_alpha(const float* __restrict__ We,
                                                    const float* __restrict__ attn,
                                                    float* __restrict__ w_alpha) {
    int idx = blockIdx.x * 256 + threadIdx.x;            // 512*8 threads
    if (idx >= D_N * H_N) return;
    int h = idx & (H_N - 1);
    int k = idx >> 3;
    float s = 0.f;
    #pragma unroll 8
    for (int c = 0; c < CD_N; ++c)
        s += We[(size_t)k * D_N + h * CD_N + c] * attn[h * (3 * CD_N) + 2 * CD_N + c];
    w_alpha[k * H_N + h] = s;
}

// ---------------- ep_alpha = edge_feats @ w_alpha (wave per edge) ----------------
__global__ __launch_bounds__(256) void k_edge_alpha(const float* __restrict__ ef,
                                                    const float* __restrict__ w_alpha,
                                                    float* __restrict__ out) {
    int gw = (blockIdx.x * 256 + threadIdx.x) >> 5;      // edge id
    int lane = threadIdx.x & 31;
    if (gw >= E_N) return;
    const float* row = ef + (size_t)gw * D_N;
    float acc[H_N] = {};
    int k = lane * 16;
    #pragma unroll 4
    for (int i = 0; i < 16; ++i, ++k) {
        float x = row[k];
        const float* wa = w_alpha + k * H_N;
        #pragma unroll
        for (int h = 0; h < H_N; ++h) acc[h] += x * wa[h];
    }
    #pragma unroll
    for (int h = 0; h < H_N; ++h) {
        float v = acc[h];
        #pragma unroll
        for (int off = 16; off > 0; off >>= 1) v += __shfl_xor(v, off, 32);
        if (lane == 0) out[(size_t)gw * H_N + h] = v;
    }
}

// ---------------- WMMA GEMM:  C[M x 512] = A[M x 512](f32) @ BT^T + bias ----------------
// BT is 512x512 f16 stored [n][k] (pre-transposed).
// Block tile 64(M) x 256(N); 8 waves as 2(M) x 4(N); each wave 32M x 64N = 8 WMMA tiles.
// B tile staged via global_load_async_to_lds_b128 (raw f16, no conversion needed);
// A tile staged through VGPRs (f32 -> f16 convert).
__global__ __launch_bounds__(256) void k_gemm_wmma(const float* __restrict__ A, int M,
                                                   const _Float16* __restrict__ BT,
                                                   const float* __restrict__ bias,
                                                   float* __restrict__ C) {
    __shared__ _Float16 lA[64 * 32];     // [row][k]   4 KB
    __shared__ _Float16 lB[256 * 32];    // [n][k]    16 KB
    const int tid  = threadIdx.x;
    const int lane = tid & 31;
    const int w    = tid >> 5;
    const int wm   = w & 1;             // 2 M sub-tiles (32 rows each)
    const int wn   = w >> 1;            // 4 N sub-tiles (64 cols each)
    const int bm   = blockIdx.x * 64;
    const int bn   = blockIdx.y * 256;

    v8f acc[2][4] = {};

    for (int k0 = 0; k0 < D_N; k0 += 32) {
        // ---- stage B tile (256 cols x 32 k) via async copy: 4 x b128 per thread ----
        {
            const _Float16* gb = BT + (size_t)(bn + tid) * D_N + k0;
            unsigned lds = (unsigned)(uintptr_t)&lB[tid * 32];
            #pragma unroll
            for (int j = 0; j < 4; ++j)
                async_copy_b128(lds + j * 16, gb + j * 8);
        }
        // ---- stage A tile (64x32 f32 -> f16), 8 elems/thread ----
        {
            int r  = tid >> 2;
            int cg = (tid & 3) * 8;
            int gr = bm + r;
            float4 f0 = make_float4(0.f, 0.f, 0.f, 0.f);
            float4 f1 = make_float4(0.f, 0.f, 0.f, 0.f);
            if (gr < M) {
                const float* p = A + (size_t)gr * D_N + k0 + cg;
                f0 = *reinterpret_cast<const float4*>(p);
                f1 = *reinterpret_cast<const float4*>(p + 4);
                if (k0 + 32 < D_N)
                    __builtin_prefetch(p + 32, 0, 0);    // global_prefetch_b8
            }
            _Float16* d = &lA[r * 32 + cg];
            d[0] = (_Float16)f0.x; d[1] = (_Float16)f0.y;
            d[2] = (_Float16)f0.z; d[3] = (_Float16)f0.w;
            d[4] = (_Float16)f1.x; d[5] = (_Float16)f1.y;
            d[6] = (_Float16)f1.z; d[7] = (_Float16)f1.w;
        }
        async_wait0();                   // B tile landed in LDS
        __syncthreads();                 // all waves' stores visible

        // ---- A fragments (16x32 layout: lanes 0-15 K{0..7,16..23}; 16-31 K{8..15,24..31}) ----
        HF16x16 af[2];
        #pragma unroll
        for (int mi = 0; mi < 2; ++mi) {
            int row = wm * 32 + mi * 16 + (lane & 15);
            int kb  = (lane < 16) ? 0 : 8;
            const _Float16* pr = &lA[row * 32 + kb];
            #pragma unroll
            for (int j = 0; j < 8; ++j) {
                int kk = (j < 4) ? (2 * j) : (16 + 2 * (j - 4));
                af[mi].h[2 * j]     = pr[kk];
                af[mi].h[2 * j + 1] = pr[kk + 1];
            }
        }
        // ---- B fragments + 8 WMMAs ----
        #pragma unroll
        for (int s = 0; s < 4; ++s) {
            HF16x16 bf;
            int n  = wn * 64 + s * 16 + (lane & 15);
            int kh = (lane < 16) ? 0 : 8;
            const _Float16* pb = &lB[n * 32 + kh];
            #pragma unroll
            for (int i = 0; i < 16; ++i) bf.h[i] = pb[(i < 8) ? i : (i + 8)];
            #pragma unroll
            for (int mi = 0; mi < 2; ++mi)
                acc[mi][s] = __builtin_amdgcn_wmma_f32_16x16x32_f16(
                    false, af[mi].v, false, bf.v, (short)0, acc[mi][s], false, false);
        }
        __syncthreads();
    }

    // ---- epilogue: lane n = lane&15; VGPR r -> row r (+8 for lanes 16-31) ----
    #pragma unroll
    for (int mi = 0; mi < 2; ++mi) {
        int rbase = bm + wm * 32 + mi * 16 + ((lane < 16) ? 0 : 8);
        #pragma unroll
        for (int s = 0; s < 4; ++s) {
            int col = bn + wn * 64 + s * 16 + (lane & 15);
            float b = bias ? bias[col] : 0.f;
            #pragma unroll
            for (int r = 0; r < 8; ++r) {
                int row = rbase + r;
                if (row < M) C[(size_t)row * D_N + col] = acc[mi][s][r] + b;
            }
        }
    }
}

// ---------------- aj/ai from xp (thread per (v,h), v in [0,10000)) ----------------
__global__ __launch_bounds__(256) void k_head_dots(const float* __restrict__ xp,
                                                   const float* __restrict__ attn,
                                                   float* __restrict__ aj,
                                                   float* __restrict__ ai) {
    int idx = blockIdx.x * 256 + threadIdx.x;
    if (idx >= 2 * PER_N * H_N) return;
    int h = idx & (H_N - 1);
    int v = idx >> 3;
    const float* row = xp + (size_t)v * D_N + h * CD_N;
    const float* wj  = attn + h * (3 * CD_N);
    const float* wi  = wj + CD_N;
    float sj = 0.f, si = 0.f;
    #pragma unroll 8
    for (int c = 0; c < CD_N; ++c) { float x = row[c]; sj += x * wj[c]; si += x * wi[c]; }
    aj[idx] = sj;
    ai[idx] = si;
}

// ---------------- init m = sa (ordered int), recv nodes ----------------
__global__ __launch_bounds__(256) void k_init_m(const float* __restrict__ aj,
                                                const float* __restrict__ ai,
                                                int recvrel, int* __restrict__ mint) {
    int idx = blockIdx.x * 256 + threadIdx.x;            // 5000*8
    if (idx >= PER_N * H_N) return;
    int vr = (idx >> 3) + recvrel;
    int h  = idx & (H_N - 1);
    float sa = (aj[vr * H_N + h] + ai[vr * H_N + h]) * SCALE_A;
    mint[idx] = orderF(sa);
}

// ---------------- edge pass 1: a_e + atomic segment max ----------------
__global__ __launch_bounds__(256) void k_edge_pass1(const int* __restrict__ src_ids,
                                                    const int* __restrict__ tgt_ids,
                                                    const float* __restrict__ aj,
                                                    const float* __restrict__ ai,
                                                    const float* __restrict__ ep_alpha,
                                                    int eb, int rowbase, int recvbase,
                                                    float* __restrict__ a_e,
                                                    int* __restrict__ mint) {
    int idx = blockIdx.x * 256 + threadIdx.x;            // PB*8
    if (idx >= PB_N * H_N) return;
    int h = idx & (H_N - 1);
    int e = idx >> 3;
    int ge = eb + e;
    int s = src_ids[ge], t = tgt_ids[ge];
    float a = (aj[(s - rowbase) * H_N + h] + ai[(t - rowbase) * H_N + h]
               + ep_alpha[(size_t)ge * H_N + h]) * SCALE_A;
    a_e[e * H_N + h] = a;
    atomicMax(&mint[(t - recvbase) * H_N + h], orderF(a));
}

// ---------------- self contribution: denom=ws, agg=ws*xp ----------------
__global__ __launch_bounds__(256) void k_self_init(const float* __restrict__ xp,
                                                   const float* __restrict__ aj,
                                                   const float* __restrict__ ai,
                                                   const int* __restrict__ mint,
                                                   int recvrel,
                                                   float* __restrict__ agg,
                                                   float* __restrict__ denom) {
    int idx = blockIdx.x * 256 + threadIdx.x;            // 5000*512
    if (idx >= PER_N * D_N) return;
    int d = idx & (D_N - 1);
    int v = idx >> 9;
    int h = d >> 6;
    int vr = v + recvrel;
    float m  = unorderF(mint[v * H_N + h]);
    float sa = (aj[vr * H_N + h] + ai[vr * H_N + h]) * SCALE_A;
    float ws = __expf(sa - m);
    agg[idx] = ws * xp[(size_t)vr * D_N + d];
    if ((d & 63) == 0) denom[v * H_N + h] = ws;
}

// ---------------- edge scatter: atomicAdd agg / denom ----------------
__global__ __launch_bounds__(256) void k_edge_scatter(const int* __restrict__ src_ids,
                                                      const int* __restrict__ tgt_ids,
                                                      const float* __restrict__ a_e,
                                                      const int* __restrict__ mint,
                                                      const float* __restrict__ xp,
                                                      int eb, int rowbase, int recvbase,
                                                      float* __restrict__ agg,
                                                      float* __restrict__ denom) {
    int e  = blockIdx.x;                                 // PB blocks
    int ge = eb + e;
    int s  = src_ids[ge], t = tgt_ids[ge];
    int sr = s - rowbase, tv = t - recvbase;
    #pragma unroll
    for (int j = 0; j < 2; ++j) {
        int d = threadIdx.x + j * 256;
        int h = d >> 6;
        float we = __expf(a_e[e * H_N + h] - unorderF(mint[tv * H_N + h]));
        atomicAdd(&agg[(size_t)tv * D_N + d], we * xp[(size_t)sr * D_N + d]);
        if ((d & 63) == 0) atomicAdd(&denom[tv * H_N + h], we);
    }
}

// ---------------- normalize agg /= denom ----------------
__global__ __launch_bounds__(256) void k_normalize(float* __restrict__ agg,
                                                   const float* __restrict__ denom) {
    int idx = blockIdx.x * 256 + threadIdx.x;            // 5000*512
    if (idx >= PER_N * D_N) return;
    int d = idx & (D_N - 1);
    int v = idx >> 9;
    agg[idx] /= denom[v * H_N + (d >> 6)];
}

// =====================================================================
struct WS {
    float*    xp;       // 10000 x 512
    float*    aj;       // 10000 x 8
    float*    ai;       // 10000 x 8
    float*    ep_alpha; // 240000 x 8
    float*    a_e;      // 80000 x 8
    int*      mint;     // 5000 x 8
    float*    denom;    // 5000 x 8
    float*    agg;      // 5000 x 512
    float*    w_alpha;  // 512 x 8
    _Float16* WnT;      // 512 x 512
    _Float16* WoT;      // 512 x 512
};

static void run_direction(const float* edge_feats, const int* src_ids, const int* tgt_ids,
                          const float* Wn, const float* We, const float* attn,
                          const float* Wo, const float* bo,
                          float* cur, bool bu, const WS& W, hipStream_t stream) {
    // weight prep + folded edge-alpha
    k_cvt_transpose<<<(D_N * D_N + 255) / 256, 256, 0, stream>>>(Wn, W.WnT);
    k_cvt_transpose<<<(D_N * D_N + 255) / 256, 256, 0, stream>>>(Wo, W.WoT);
    k_fold_alpha<<<(D_N * H_N + 255) / 256, 256, 0, stream>>>(We, attn, W.w_alpha);
    k_edge_alpha<<<(E_N * 32 + 255) / 256, 256, 0, stream>>>(edge_feats, W.w_alpha, W.ep_alpha);

    for (int l = 1; l < L_N; ++l) {
        int recvbase = (bu ? l : (3 - l)) * PER_N;
        int sendbase = (bu ? (l - 1) : (4 - l)) * PER_N;
        int rowbase  = (recvbase < sendbase) ? recvbase : sendbase;
        int eb       = (bu ? (l - 1) : (3 - l)) * PB_N;
        int recvrel  = recvbase - rowbase;

        // xp for sender+receiver levels (contiguous 10000 rows)
        k_gemm_wmma<<<dim3((2 * PER_N + 63) / 64, D_N / 256), 256, 0, stream>>>(
            cur + (size_t)rowbase * D_N, 2 * PER_N, W.WnT, nullptr, W.xp);
        k_head_dots<<<(2 * PER_N * H_N + 255) / 256, 256, 0, stream>>>(W.xp, attn, W.aj, W.ai);
        k_init_m<<<(PER_N * H_N + 255) / 256, 256, 0, stream>>>(W.aj, W.ai, recvrel, W.mint);
        k_edge_pass1<<<(PB_N * H_N + 255) / 256, 256, 0, stream>>>(
            src_ids, tgt_ids, W.aj, W.ai, W.ep_alpha, eb, rowbase, recvbase, W.a_e, W.mint);
        k_self_init<<<(PER_N * D_N + 255) / 256, 256, 0, stream>>>(
            W.xp, W.aj, W.ai, W.mint, recvrel, W.agg, W.denom);
        k_edge_scatter<<<PB_N, 256, 0, stream>>>(
            src_ids, tgt_ids, W.a_e, W.mint, W.xp, eb, rowbase, recvbase, W.agg, W.denom);
        k_normalize<<<(PER_N * D_N + 255) / 256, 256, 0, stream>>>(W.agg, W.denom);
        // out = agg @ Wo + bo  -> write into cur[recv rows]
        k_gemm_wmma<<<dim3((PER_N + 63) / 64, D_N / 256), 256, 0, stream>>>(
            W.agg, PER_N, W.WoT, bo, cur + (size_t)recvbase * D_N);
    }
}

extern "C" void kernel_launch(void* const* d_in, const int* in_sizes, int n_in,
                              void* d_out, int out_size, void* d_ws, size_t ws_size,
                              hipStream_t stream) {
    (void)in_sizes; (void)n_in; (void)out_size; (void)ws_size;

    const float* node_feats = (const float*)d_in[0];
    const float* edge_feats = (const float*)d_in[1];
    const int*   eidx       = (const int*)d_in[2];
    const float* Wn_bu = (const float*)d_in[5];
    const float* We_bu = (const float*)d_in[6];
    const float* at_bu = (const float*)d_in[7];
    const float* Wo_bu = (const float*)d_in[8];
    const float* bo_bu = (const float*)d_in[9];
    const float* Wn_td = (const float*)d_in[10];
    const float* We_td = (const float*)d_in[11];
    const float* at_td = (const float*)d_in[12];
    const float* Wo_td = (const float*)d_in[13];
    const float* bo_td = (const float*)d_in[14];

    float* out_td = (float*)d_out;                       // stack order: [td, bu]
    float* out_bu = out_td + (size_t)V_N * D_N;

    // carve workspace
    char* p = (char*)d_ws;
    size_t off = 0;
    auto carve = [&](size_t bytes) -> void* {
        void* q = p + off;
        off = (off + bytes + 255) & ~(size_t)255;
        return q;
    };
    WS W;
    W.xp       = (float*)carve((size_t)2 * PER_N * D_N * 4);
    W.aj       = (float*)carve((size_t)2 * PER_N * H_N * 4);
    W.ai       = (float*)carve((size_t)2 * PER_N * H_N * 4);
    W.ep_alpha = (float*)carve((size_t)E_N * H_N * 4);
    W.a_e      = (float*)carve((size_t)PB_N * H_N * 4);
    W.mint     = (int*)  carve((size_t)PER_N * H_N * 4);
    W.denom    = (float*)carve((size_t)PER_N * H_N * 4);
    W.agg      = (float*)carve((size_t)PER_N * D_N * 4);
    W.w_alpha  = (float*)carve((size_t)D_N * H_N * 4);
    W.WnT      = (_Float16*)carve((size_t)D_N * D_N * 2);
    W.WoT      = (_Float16*)carve((size_t)D_N * D_N * 2);

    // init both direction buffers with original node feats
    hipMemcpyAsync(out_td, node_feats, (size_t)V_N * D_N * 4, hipMemcpyDeviceToDevice, stream);
    hipMemcpyAsync(out_bu, node_feats, (size_t)V_N * D_N * 4, hipMemcpyDeviceToDevice, stream);

    const int* src_all = eidx;
    const int* tgt_all = eidx + E_N;

    // bottom-up: messages src->tgt ; top-down: swapped roles
    run_direction(edge_feats, src_all, tgt_all, Wn_bu, We_bu, at_bu, Wo_bu, bo_bu,
                  out_bu, /*bu=*/true, W, stream);
    run_direction(edge_feats, tgt_all, src_all, Wn_td, We_td, at_td, Wo_td, bo_td,
                  out_td, /*bu=*/false, W, stream);
}
